// Model_81990925681026
// MI455X (gfx1250) — compile-verified, matrix operand order checked by reading, not run on hardware
//
#include <hip/hip_runtime.h>
#include <hip/hip_bf16.h>

// ---------------- model constants ----------------
#define B_      8
#define L_      256
#define N_      1024
#define M_      8
#define E_      16384
#define C_      50
#define EMB_    768
#define D_      256
#define TDIM_   50
#define H_      4
#define DH_     64
#define KSPLIT_ 512
#define NN_     (N_ + 1)       // 1025 nodes per graph
#define LOGROWS (KSPLIT_ * M_) // 4096 gathered logic rows per batch
#define HID_    1024

typedef __bf16 bf16x8  __attribute__((ext_vector_type(8)));
typedef __bf16 bf16x16 __attribute__((ext_vector_type(16)));
typedef float  v8f     __attribute__((ext_vector_type(8)));

static __device__ __forceinline__ unsigned fmap(float f) {
    unsigned u = __float_as_uint(f);
    return (u & 0x80000000u) ? ~u : (u | 0x80000000u);
}
static __device__ __forceinline__ float funmap(unsigned u) {
    u = (u & 0x80000000u) ? (u & 0x7FFFFFFFu) : ~u;
    return __uint_as_float(u);
}

// =====================================================================
// Generic tiled GEMM:  C[M,N] = op(A)[M,K] @ B[K,N] (+ bias[col]*rowscale)
// op(A): optional row gather (a_idx) for the entity-table MLP.
// f32 in memory, bf16 in LDS/registers, f32 WMMA accumulate.
// Block = 256 threads = 8 waves; 128x64 block tile; K-step 32.
// Waves arranged 4 (M, 32 rows each) x 2 (N, 32 cols each);
// each wave does a 2x2 grid of 16x16 WMMA tiles (4 WMMAs / K-step).
// LDS double-buffered; software-pipelined staging.
//
// Out-of-range A rows / B cols are NOT zeroed (addresses are clamped to
// valid memory): a bad row/col only pollutes accumulator elements whose
// store is bounds-guarded anyway. Only the K tail needs zeroing, so the
// FULLK=true instantiation (K % 32 == 0; all large GEMMs here) has a
// completely mask-free staging path with b128 global loads.
// =====================================================================
template <bool FULLK>
__global__ __launch_bounds__(256) void wmma_gemm_bf16(
    const float* __restrict__ A, const int* __restrict__ a_idx, int lda,
    const float* __restrict__ Bw, int ldb,
    const float* __restrict__ bias, const float* __restrict__ bias_rowscale,
    float* __restrict__ C, int ldc, int M, int N, int K, int relu)
{
    __shared__ __bf16 As[2][128][40];  // pitch 40 (80 B) keeps 16 B alignment
    __shared__ __bf16 Bs[2][64][40];

    const int tid  = threadIdx.x;
    const int lane = tid & 31;
    const int wave = tid >> 5;
    const int waveM = (wave & 3) * 32;
    const int waveN = (wave >> 2) * 32;
    const int lr = lane & 15;   // row/col within a 16x16 tile
    const int lh = lane >> 4;   // lane half -> K sub-block select
    const int rowBase = blockIdx.y * 128;
    const int colBase = blockIdx.x * 64;

    // ---- A staging assignment: 2 chunks of 8 per thread (hoisted) ----
    int rA[2], cA[2];
    const float* rpA[2];
#pragma unroll
    for (int i = 0; i < 2; ++i) {
        int ch = tid + i * 256;          // 512 chunks: 128 rows x 4 chunks
        int r  = ch >> 2;
        int c8 = (ch & 3) * 8;
        int gr = rowBase + r;
        bool ok = gr < M;
        long ridx = ok ? (a_idx ? (long)a_idx[gr] : (long)gr) : 0;  // clamp
        rpA[i] = A + (size_t)ridx * (size_t)lda;
        rA[i] = r; cA[i] = c8;
    }
    // ---- B staging assignment: one column, 8 K-rows per thread ----
    const int cB   = tid & 63;
    const int krB  = (tid >> 6) * 8;
    const int gcB  = colBase + cB;
    const int gcBc = (gcB < N) ? gcB : 0;  // clamped column (always valid)

    float fA[2][8];
    float fB[8];

    // global loads of one K-tile into registers (mask-free when FULLK)
    auto gload = [&](int kb) {
#pragma unroll
        for (int i = 0; i < 2; ++i) {
            int kc = kb + cA[i];
            if (FULLK) {
                const float4* p = (const float4*)(rpA[i] + kc); // 16B-aligned for all lda used
                float4 t0 = p[0], t1 = p[1];
                fA[i][0] = t0.x; fA[i][1] = t0.y; fA[i][2] = t0.z; fA[i][3] = t0.w;
                fA[i][4] = t1.x; fA[i][5] = t1.y; fA[i][6] = t1.z; fA[i][7] = t1.w;
            } else {
#pragma unroll
                for (int j = 0; j < 8; ++j) {
                    int gk = kc + j;
                    fA[i][j] = rpA[i][(gk < K) ? gk : 0];
                }
            }
        }
#pragma unroll
        for (int j = 0; j < 8; ++j) {
            int gk = kb + krB + j;
            if (!FULLK) gk = (gk < K) ? gk : 0;
            fB[j] = Bw[(size_t)gk * ldb + gcBc];
        }
    };
    // convert + b128 LDS stores (zero K-tail only in non-FULLK)
    auto lstore = [&](int bufi, int kb) {
#pragma unroll
        for (int i = 0; i < 2; ++i) {
            int kc = kb + cA[i];
            bf16x8 pack;
#pragma unroll
            for (int j = 0; j < 8; ++j) {
                float v = fA[i][j];
                if (!FULLK && kc + j >= K) v = 0.f;
                pack[j] = (__bf16)v;
            }
            *(bf16x8*)&As[bufi][rA[i]][cA[i]] = pack;
        }
        {
            bf16x8 pack;
#pragma unroll
            for (int j = 0; j < 8; ++j) {
                float v = fB[j];
                if (!FULLK && kb + krB + j >= K) v = 0.f;
                pack[j] = (__bf16)v;
            }
            *(bf16x8*)&Bs[bufi][cB][krB] = pack;
        }
    };

    v8f acc00 = {}, acc01 = {}, acc10 = {}, acc11 = {};

    const int nk = (K + 31) >> 5;
    gload(0);
    lstore(0, 0);
    __syncthreads();

    int cur = 0;
    for (int t = 0; t < nk; ++t) {
        const int kbn = (t + 1) << 5;
        if (t + 1 < nk) gload(kbn);          // long-latency loads first

        // ---- fragments: two contiguous b128 LDS reads each ----
        const int k0 = lh * 8, k1 = 16 + lh * 8;
        bf16x8 aL0 = *(const bf16x8*)&As[cur][waveM + lr][k0];
        bf16x8 aH0 = *(const bf16x8*)&As[cur][waveM + lr][k1];
        bf16x8 aL1 = *(const bf16x8*)&As[cur][waveM + 16 + lr][k0];
        bf16x8 aH1 = *(const bf16x8*)&As[cur][waveM + 16 + lr][k1];
        bf16x8 bL0 = *(const bf16x8*)&Bs[cur][waveN + lr][k0];
        bf16x8 bH0 = *(const bf16x8*)&Bs[cur][waveN + lr][k1];
        bf16x8 bL1 = *(const bf16x8*)&Bs[cur][waveN + 16 + lr][k0];
        bf16x8 bH1 = *(const bf16x8*)&Bs[cur][waveN + 16 + lr][k1];
        bf16x16 a0 = __builtin_shufflevector(aL0, aH0, 0,1,2,3,4,5,6,7,8,9,10,11,12,13,14,15);
        bf16x16 a1 = __builtin_shufflevector(aL1, aH1, 0,1,2,3,4,5,6,7,8,9,10,11,12,13,14,15);
        bf16x16 b0 = __builtin_shufflevector(bL0, bH0, 0,1,2,3,4,5,6,7,8,9,10,11,12,13,14,15);
        bf16x16 b1 = __builtin_shufflevector(bL1, bH1, 0,1,2,3,4,5,6,7,8,9,10,11,12,13,14,15);

        acc00 = __builtin_amdgcn_wmma_f32_16x16x32_bf16(false, a0, false, b0, (short)0, acc00, false, false);
        acc01 = __builtin_amdgcn_wmma_f32_16x16x32_bf16(false, a0, false, b1, (short)0, acc01, false, false);
        acc10 = __builtin_amdgcn_wmma_f32_16x16x32_bf16(false, a1, false, b0, (short)0, acc10, false, false);
        acc11 = __builtin_amdgcn_wmma_f32_16x16x32_bf16(false, a1, false, b1, (short)0, acc11, false, false);

        if (t + 1 < nk) lstore(cur ^ 1, kbn); // waits for loads AFTER the WMMAs
        __syncthreads();
        cur ^= 1;
    }

    // ---- store (f32 C/D layout: VGPR j -> M = j + lh*8) ----
#pragma unroll
    for (int mt = 0; mt < 2; ++mt) {
#pragma unroll
        for (int j = 0; j < 8; ++j) {
            int r = rowBase + waveM + mt * 16 + lh * 8 + j;
            if (r >= M) continue;
            float rs = bias_rowscale ? bias_rowscale[r] : 1.f;
            int c0 = colBase + waveN + lr;
            int c1 = c0 + 16;
            float x0 = (mt == 0 ? acc00[j] : acc10[j]);
            float x1 = (mt == 0 ? acc01[j] : acc11[j]);
            if (c0 < N) {
                float x = x0 + (bias ? bias[c0] * rs : 0.f);
                if (relu) x = fmaxf(x, 0.f);
                C[(size_t)r * ldc + c0] = x;
            }
            if (c1 < N) {
                float x = x1 + (bias ? bias[c1] * rs : 0.f);
                if (relu) x = fmaxf(x, 0.f);
                C[(size_t)r * ldc + c1] = x;
            }
        }
    }
}

// =====================================================================
// text_vec[b,e] = sum_l mask*word_embed[sentence[b,l],e] / (sum mask + eps)
// =====================================================================
__global__ void text_pool_kernel(const int* __restrict__ sent,
                                 const float* __restrict__ mask,
                                 const float* __restrict__ wemb,
                                 float* __restrict__ tv)
{
    int idx = blockIdx.x * blockDim.x + threadIdx.x;
    if (idx >= B_ * EMB_) return;
    int b = idx / EMB_, e = idx % EMB_;
    float acc = 0.f, ms = 0.f;
    for (int l = 0; l < L_; ++l) {
        float m = mask[b * L_ + l];
        ms += m;
        acc += m * wemb[(size_t)sent[b * L_ + l] * EMB_ + e];
    }
    tv[idx] = acc / (ms + 1e-9f);
}

// KG half: nepool[n,e] = sum_m mask * entity_table[nodes[n,m], e]   (n < 512)
__global__ void kg_pool_kernel(const int* __restrict__ nodes_b,
                               const float* __restrict__ nmask_b,
                               const float* __restrict__ etab,
                               float* __restrict__ nepool)
{
    int idx = blockIdx.x * blockDim.x + threadIdx.x;
    if (idx >= KSPLIT_ * EMB_) return;
    int n = idx / EMB_, e = idx % EMB_;
    float acc = 0.f;
    for (int m = 0; m < M_; ++m) {
        float mm = nmask_b[n * M_ + m];
        acc += mm * etab[(size_t)nodes_b[n * M_ + m] * EMB_ + e];
    }
    nepool[idx] = acc;
}

// Pool relu'd hidden over M=8 (linear after relu); msum[n] = sum_m mask
__global__ void hid_pool_kernel(const float* __restrict__ hid,
                                const float* __restrict__ nmask_logic,
                                float* __restrict__ hpool,
                                float* __restrict__ msum)
{
    int idx = blockIdx.x * blockDim.x + threadIdx.x;
    if (idx >= KSPLIT_ * HID_) return;
    int n = idx / HID_, k = idx % HID_;
    float acc = 0.f, ms = 0.f;
    for (int m = 0; m < M_; ++m) {
        float mm = nmask_logic[n * M_ + m];
        ms += mm;
        acc += mm * hid[(size_t)(n * M_ + m) * HID_ + k];
    }
    hpool[idx] = acc;
    if (k == 0) msum[n] = ms;
}

// hsrc = [text_vec_b ; ne2]  (1025 x 768)
__global__ void assemble_hsrc_kernel(const float* __restrict__ tv_b,
                                     const float* __restrict__ ne2,
                                     float* __restrict__ hsrc)
{
    int idx = blockIdx.x * blockDim.x + threadIdx.x;
    if (idx >= NN_ * EMB_) return;
    int r = idx / EMB_, c = idx % EMB_;
    hsrc[idx] = (r == 0) ? tv_b[c] : ne2[(size_t)(r - 1) * EMB_ + c];
}

// h[i] += (type_table @ type_proj_w)[node_types[i]]
__global__ void add_type_kernel(float* __restrict__ h,
                                const float* __restrict__ ttp,
                                const int* __restrict__ nt_b)
{
    int idx = blockIdx.x * blockDim.x + threadIdx.x;
    if (idx >= NN_ * D_) return;
    int r = idx / D_, c = idx % D_;
    h[idx] += ttp[(size_t)nt_b[r] * D_ + c];
}

// reset per-layer attention scratch
__global__ void gnn_init_kernel(unsigned* __restrict__ mxu,
                                float* __restrict__ den,
                                float* __restrict__ agg)
{
    int idx = blockIdx.x * blockDim.x + threadIdx.x;
    if (idx < NN_ * H_) { mxu[idx] = 0u; den[idx] = 0.f; }
    if (idx < NN_ * D_) agg[idx] = 0.f;
}

// logits[e,h] = <q[dst], k[src]+ke[type]> / 8 ; segment max via mapped atomicMax
__global__ void edge_logits_kernel(const float* __restrict__ q,
                                   const float* __restrict__ k,
                                   const float* __restrict__ ketab,
                                   const int* __restrict__ src,
                                   const int* __restrict__ dst,
                                   const int* __restrict__ et,
                                   float* __restrict__ logits,
                                   unsigned* __restrict__ mxu)
{
    int idx = blockIdx.x * blockDim.x + threadIdx.x;
    if (idx >= E_ * H_) return;
    int e = idx >> 2, hh = idx & 3;
    int s = src[e], d = dst[e], t = et[e];
    const float* qp  = q + (size_t)d * D_ + hh * DH_;
    const float* kp  = k + (size_t)s * D_ + hh * DH_;
    const float* kep = ketab + (size_t)t * D_ + hh * DH_;
    float acc = 0.f;
#pragma unroll 8
    for (int j = 0; j < DH_; ++j) acc += qp[j] * (kp[j] + kep[j]);
    acc *= 0.125f;  // 1/sqrt(64)
    logits[idx] = acc;
    atomicMax(&mxu[d * H_ + hh], fmap(acc));
}

// ex = exp(logit - mx[dst]); segment sum into den
__global__ void edge_exp_kernel(const int* __restrict__ dst,
                                float* __restrict__ logits,
                                const unsigned* __restrict__ mxu,
                                float* __restrict__ den)
{
    int idx = blockIdx.x * blockDim.x + threadIdx.x;
    if (idx >= E_ * H_) return;
    int e = idx >> 2, hh = idx & 3;
    int d = dst[e];
    float ex = __expf(logits[idx] - funmap(mxu[d * H_ + hh]));
    logits[idx] = ex;
    atomicAdd(&den[d * H_ + hh], ex);
}

// agg[dst] += (ex/den) * v[src]
__global__ void edge_agg_kernel(const int* __restrict__ src,
                                const int* __restrict__ dst,
                                const float* __restrict__ logits,
                                const float* __restrict__ den,
                                const float* __restrict__ v,
                                float* __restrict__ agg)
{
    int idx = blockIdx.x * blockDim.x + threadIdx.x;
    if (idx >= E_ * H_) return;
    int e = idx >> 2, hh = idx & 3;
    int s = src[e], d = dst[e];
    float alpha = logits[idx] / (den[d * H_ + hh] + 1e-9f);
    const float* vp = v + (size_t)s * D_ + hh * DH_;
    float* ap = agg + (size_t)d * D_ + hh * DH_;
#pragma unroll 4
    for (int j = 0; j < DH_; ++j) atomicAdd(&ap[j], alpha * vp[j]);
}

__global__ void h_update_kernel(float* __restrict__ h, const float* __restrict__ agg)
{
    int idx = blockIdx.x * blockDim.x + threadIdx.x;
    if (idx >= NN_ * D_) return;
    h[idx] = fmaxf(h[idx] + agg[idx], 0.f);
}

// pool[c] = [ h[0] (ctx) | h[1+c] (label) | text_vec_b (prompt) ]  -> 50 x 1280
__global__ void assemble_pool_kernel(const float* __restrict__ h,
                                     const float* __restrict__ tv_b,
                                     float* __restrict__ poolm)
{
    int idx = blockIdx.x * blockDim.x + threadIdx.x;
    if (idx >= C_ * (EMB_ + 2 * D_)) return;
    int c = idx / (EMB_ + 2 * D_), j = idx % (EMB_ + 2 * D_);
    float val;
    if (j < D_)            val = h[j];
    else if (j < 2 * D_)   val = h[(size_t)(1 + c) * D_ + (j - D_)];
    else                   val = tv_b[j - 2 * D_];
    poolm[idx] = val;
}

// out[c] = <poolout[c], cls2_w[c]> + cls2_b[c]
__global__ void rowdot_kernel(const float* __restrict__ A,
                              const float* __restrict__ Bm,
                              const float* __restrict__ bias,
                              float* __restrict__ out, int K)
{
    int c = blockIdx.x;
    float s = 0.f;
    for (int j = threadIdx.x; j < K; j += 256)
        s += A[(size_t)c * K + j] * Bm[(size_t)c * K + j];
    __shared__ float red[256];
    red[threadIdx.x] = s;
    __syncthreads();
    for (int st = 128; st > 0; st >>= 1) {
        if (threadIdx.x < st) red[threadIdx.x] += red[threadIdx.x + st];
        __syncthreads();
    }
    if (threadIdx.x == 0) out[c] = red[0] + bias[c];
}

// =====================================================================
static inline int cdiv(int a, int b) { return (a + b - 1) / b; }

static inline void gemm(hipStream_t st, const float* A, const int* aidx, int lda,
                        const float* Bw, int ldb, const float* bias,
                        const float* brs, float* C, int ldc,
                        int M, int N, int K, int relu)
{
    dim3 grid(cdiv(N, 64), cdiv(M, 128));
    if ((K & 31) == 0)
        wmma_gemm_bf16<true><<<grid, 256, 0, st>>>(A, aidx, lda, Bw, ldb, bias, brs,
                                                   C, ldc, M, N, K, relu);
    else
        wmma_gemm_bf16<false><<<grid, 256, 0, st>>>(A, aidx, lda, Bw, ldb, bias, brs,
                                                    C, ldc, M, N, K, relu);
}

extern "C" void kernel_launch(void* const* d_in, const int* in_sizes, int n_in,
                              void* d_out, int out_size, void* d_ws, size_t ws_size,
                              hipStream_t stream)
{
    const int*   sentence     = (const int*)  d_in[0];
    const float* mask         = (const float*)d_in[1];
    const int*   nodes        = (const int*)  d_in[2];
    const float* node_mask    = (const float*)d_in[3];
    const int*   node_types   = (const int*)  d_in[4];
    const int*   edges        = (const int*)  d_in[5];
    const int*   edge_types   = (const int*)  d_in[6];
    const float* word_embed   = (const float*)d_in[7];
    const float* entity_table = (const float*)d_in[8];
    const float* type_table   = (const float*)d_in[9];
    const float* mlp_w1       = (const float*)d_in[10];
    const float* mlp_b1       = (const float*)d_in[11];
    const float* mlp_w2       = (const float*)d_in[12];
    const float* mlp_b2       = (const float*)d_in[13];
    const float* ntw_w        = (const float*)d_in[14];
    const float* proj_w       = (const float*)d_in[15];
    const float* proj_b       = (const float*)d_in[16];
    const float* type_proj_w  = (const float*)d_in[17];
    const float* edge_type_tb = (const float*)d_in[18];
    const float* gnn_wq       = (const float*)d_in[19];
    const float* gnn_wk       = (const float*)d_in[20];
    const float* gnn_wv       = (const float*)d_in[21];
    const float* gnn_we       = (const float*)d_in[22];
    const float* cls_w        = (const float*)d_in[23];
    const float* cls_b        = (const float*)d_in[24];
    const float* cls2_w       = (const float*)d_in[25];
    const float* cls2_b       = (const float*)d_in[26];
    float* out = (float*)d_out;

    // ------- workspace carve-up (floats); per-batch buffers reused -------
    float* w = (float*)d_ws;
    size_t o = 0;
    float* tv     = w + o; o += (size_t)B_ * EMB_;        // text_vec [B,768]
    float* ttp    = w + o; o += (size_t)100 * D_;         // type_table@type_proj_w
    float* hid    = w + o; o += (size_t)LOGROWS * HID_;   // [4096,1024]
    float* hpool  = w + o; o += (size_t)KSPLIT_ * HID_;   // [512,1024]
    float* msum   = w + o; o += KSPLIT_;
    float* nepool = w + o; o += (size_t)N_ * EMB_;        // [1024,768]
    float* ne2    = w + o; o += (size_t)N_ * EMB_;
    float* hsrc   = w + o; o += (size_t)NN_ * EMB_;       // [1025,768]
    float* hbuf   = w + o; o += (size_t)NN_ * D_;         // [1025,256]
    float* qb     = w + o; o += (size_t)NN_ * D_;
    float* kb     = w + o; o += (size_t)NN_ * D_;
    float* vb     = w + o; o += (size_t)NN_ * D_;
    float* ketab  = w + o; o += (size_t)TDIM_ * D_;       // [50,256]
    float* logit  = w + o; o += (size_t)E_ * H_;          // [16384,4]
    unsigned* mxu = (unsigned*)(w + o); o += (size_t)NN_ * H_;
    float* den    = w + o; o += (size_t)NN_ * H_;
    float* agg    = w + o; o += (size_t)NN_ * D_;
    float* poolm  = w + o; o += (size_t)C_ * (EMB_ + 2 * D_);
    float* poolo  = w + o; o += (size_t)C_ * EMB_;
    (void)ws_size; (void)in_sizes; (void)n_in; (void)out_size;

    const int TPB = 256;

    // text pooling (all batches)
    text_pool_kernel<<<cdiv(B_ * EMB_, TPB), TPB, 0, stream>>>(sentence, mask, word_embed, tv);

    // type_table @ type_proj_w  -> ttp [100,256]
    gemm(stream, type_table, nullptr, TDIM_, type_proj_w, D_,
         nullptr, nullptr, ttp, D_, 100, D_, TDIM_, 0);

    for (int b = 0; b < B_; ++b) {
        const int*   nodes_b = nodes + (size_t)b * N_ * M_;
        const float* nmask_b = node_mask + (size_t)b * N_ * M_;
        const int*   nt_b    = node_types + (size_t)b * NN_;
        const int*   src_b   = edges + (size_t)b * 2 * E_;
        const int*   dst_b   = src_b + E_;
        const int*   et_b    = edge_types + (size_t)b * E_;
        const float* tv_b    = tv + (size_t)b * EMB_;

        // KG half: pool-before-ntw (linear)  -> nepool rows [0,512)
        kg_pool_kernel<<<cdiv(KSPLIT_ * EMB_, TPB), TPB, 0, stream>>>(
            nodes_b, nmask_b, entity_table, nepool);

        // Logic half: mlp1 on gathered rows (WMMA, relu)
        gemm(stream, entity_table, nodes_b + KSPLIT_ * M_, EMB_,
             mlp_w1, HID_, mlp_b1, nullptr, hid, HID_, LOGROWS, HID_, EMB_, 1);

        // pool hidden over M (linear after relu), track mask sums for bias
        hid_pool_kernel<<<cdiv(KSPLIT_ * HID_, TPB), TPB, 0, stream>>>(
            hid, nmask_b + KSPLIT_ * M_, hpool, msum);

        // mlp2: hpool @ W2 + msum*b2 -> nepool rows [512,1024)
        gemm(stream, hpool, nullptr, HID_, mlp_w2, EMB_, mlp_b2, msum,
             nepool + (size_t)KSPLIT_ * EMB_, EMB_, KSPLIT_, EMB_, HID_, 0);

        // ntw: nepool @ ntw_w -> ne2
        gemm(stream, nepool, nullptr, EMB_, ntw_w, EMB_,
             nullptr, nullptr, ne2, EMB_, N_, EMB_, EMB_, 0);

        // hsrc = [tv ; ne2]; proj -> h; + type embeddings
        assemble_hsrc_kernel<<<cdiv(NN_ * EMB_, TPB), TPB, 0, stream>>>(tv_b, ne2, hsrc);
        gemm(stream, hsrc, nullptr, EMB_, proj_w, D_, proj_b, nullptr,
             hbuf, D_, NN_, D_, EMB_, 0);
        add_type_kernel<<<cdiv(NN_ * D_, TPB), TPB, 0, stream>>>(hbuf, ttp, nt_b);

        // GNN layers
        for (int l = 0; l < 2; ++l) {
            const float* wq = gnn_wq + (size_t)l * D_ * D_;
            const float* wk = gnn_wk + (size_t)l * D_ * D_;
            const float* wv = gnn_wv + (size_t)l * D_ * D_;
            const float* we = gnn_we + (size_t)l * TDIM_ * D_;

            gemm(stream, edge_type_tb, nullptr, TDIM_, we, D_,
                 nullptr, nullptr, ketab, D_, TDIM_, D_, TDIM_, 0);
            gemm(stream, hbuf, nullptr, D_, wq, D_, nullptr, nullptr, qb, D_, NN_, D_, D_, 0);
            gemm(stream, hbuf, nullptr, D_, wk, D_, nullptr, nullptr, kb, D_, NN_, D_, D_, 0);
            gemm(stream, hbuf, nullptr, D_, wv, D_, nullptr, nullptr, vb, D_, NN_, D_, D_, 0);

            gnn_init_kernel<<<cdiv(NN_ * D_, TPB), TPB, 0, stream>>>(mxu, den, agg);
            edge_logits_kernel<<<cdiv(E_ * H_, TPB), TPB, 0, stream>>>(
                qb, kb, ketab, src_b, dst_b, et_b, logit, mxu);
            edge_exp_kernel<<<cdiv(E_ * H_, TPB), TPB, 0, stream>>>(dst_b, logit, mxu, den);
            edge_agg_kernel<<<cdiv(E_ * H_, TPB), TPB, 0, stream>>>(
                src_b, dst_b, logit, den, vb, agg);
            h_update_kernel<<<cdiv(NN_ * D_, TPB), TPB, 0, stream>>>(hbuf, agg);
        }

        // classifier head
        assemble_pool_kernel<<<cdiv(C_ * (EMB_ + 2 * D_), TPB), TPB, 0, stream>>>(
            hbuf, tv_b, poolm);
        gemm(stream, poolm, nullptr, EMB_ + 2 * D_, cls_w, EMB_, cls_b, nullptr,
             poolo, EMB_, C_, EMB_, EMB_ + 2 * D_, 0);
        rowdot_kernel<<<C_, 256, 0, stream>>>(poolo, cls2_w, cls2_b, out + (size_t)b * C_, EMB_);
    }
}
